// HydraFusion_26259430048573
// MI455X (gfx1250) — matrix operand order repaired and checked by reading, not compiled, not run on hardware
//
#include <hip/hip_runtime.h>
#include <hip/hip_bf16.h>

typedef _Float16 half16 __attribute__((ext_vector_type(16)));
typedef float   float8 __attribute__((ext_vector_type(8)));

#define HH 200
#define WW 200
#define HWP 40000           // H*W
#define NPIX 80000          // B*H*W
#define EPSBN 1e-5f

// Tiled f16 layout for all streaming operands:
//   addr(pix, k, Krows) = (pix>>4)*Krows*16 + k*16 + (pix&15)
// -> a 16-pixel tile's K-column block is one contiguous Krows*32B region.

// ---------------------------------------------------------------------------
// Bilinear grid-sample + geo-channel pack, fp32 -> f16 tiled [pixblk][640][16].
// k = c*n + j for sampled channels, then geo rows (Cf*n+j), ((Cf+1)*n+j).
// Raw feature maps (~67MB total) fit in the 192MB L2, so gathers are L2 hits.
// ---------------------------------------------------------------------------
__global__ void sample_pack(const float* __restrict__ feats,
                            const float* __restrict__ grids,
                            _Float16* __restrict__ X,
                            int Cf, int n) {
    int idx = blockIdx.x * blockDim.x + threadIdx.x;
    if (idx >= NPIX * n) return;
    int pix = idx % NPIX;          // lanes -> adjacent pixels (coalesced stores)
    int j   = idx / NPIX;
    int b   = pix / HWP;
    int hw  = pix % HWP;

    const float* gb = grids + ((size_t)(b * 2) * n + j) * HWP + hw;
    float gx = gb[0];
    float gy = gb[(size_t)n * HWP];

    // torch grid_sample(align_corners=False): x_img = gx*(Wf/W) - 0.5; Wf==W here
    float xi = gx - 0.5f, yi = gy - 0.5f;
    float x0f = floorf(xi), y0f = floorf(yi);
    int x0 = (int)x0f, y0 = (int)y0f;
    float wx1 = xi - x0f, wy1 = yi - y0f;

    float w00 = (1.f - wy1) * (1.f - wx1);
    float w01 = (1.f - wy1) * wx1;
    float w10 = wy1 * (1.f - wx1);
    float w11 = wy1 * wx1;

    bool vx0 = (x0 >= 0) & (x0 < WW),     vx1 = (x0 + 1 >= 0) & (x0 + 1 < WW);
    bool vy0 = (y0 >= 0) & (y0 < HH),     vy1 = (y0 + 1 >= 0) & (y0 + 1 < HH);
    float m00 = (vx0 & vy0) ? w00 : 0.f;
    float m01 = (vx1 & vy0) ? w01 : 0.f;
    float m10 = (vx0 & vy1) ? w10 : 0.f;
    float m11 = (vx1 & vy1) ? w11 : 0.f;

    int xc0 = min(max(x0, 0), WW - 1),     xc1 = min(max(x0 + 1, 0), WW - 1);
    int yc0 = min(max(y0, 0), HH - 1),     yc1 = min(max(y0 + 1, 0), HH - 1);
    size_t o00 = (size_t)yc0 * WW + xc0;
    size_t o01 = (size_t)yc0 * WW + xc1;
    size_t o10 = (size_t)yc1 * WW + xc0;
    size_t o11 = (size_t)yc1 * WW + xc1;

    _Float16* Xb = X + (size_t)(pix >> 4) * (640 * 16) + (pix & 15);

    const float* fb = feats + (size_t)b * Cf * HWP;
    for (int c = 0; c < Cf; ++c) {
        const float* f = fb + (size_t)c * HWP;
        float v = f[o00] * m00 + f[o01] * m01 + f[o10] * m10 + f[o11] * m11;
        Xb[(size_t)(c * n + j) * 16] = (_Float16)v;
    }
    Xb[(size_t)(Cf * n + j) * 16]       = (_Float16)(gx * (1.0f / WW));
    Xb[(size_t)((Cf + 1) * n + j) * 16] = (_Float16)(gy * (1.0f / HH));
}

// base_feats [B][64][HW] fp32 -> rows 0..63 of ff buffer (tiled, Krows=128)
__global__ void pack_base(const float* __restrict__ base, _Float16* __restrict__ Xff) {
    int idx = blockIdx.x * blockDim.x + threadIdx.x;
    if (idx >= 64 * NPIX) return;
    int pix = idx % NPIX;
    int c   = idx / NPIX;
    int b   = pix / HWP, hw = pix % HWP;
    Xff[(size_t)(pix >> 4) * (128 * 16) + (size_t)c * 16 + (pix & 15)] =
        (_Float16)base[((size_t)b * 64 + c) * HWP + hw];
}

// ---------------------------------------------------------------------------
// Fold BN into conv weights; emit A-fragment-native f16 layout, chunk-major
// with `nset` interleaved weight sets so a dual-GEMM LDS image is an
// identity copy:  block id = (kc*nset + set)*4 + mt
//   A 16x32 f16 layout used here: lane<16 -> M=lane, K(h)= h
//                                    lane>=16 -> M=lane-16, K += 16
// K >= cin zero-padded.  Bias -> ebias[set*64 + o].
// ---------------------------------------------------------------------------
__global__ void fold_bn_frag(const float* __restrict__ w, const float* __restrict__ b,
                             const float* __restrict__ gamma, const float* __restrict__ beta,
                             const float* __restrict__ mean, const float* __restrict__ var,
                             _Float16* __restrict__ Wfrag, float* __restrict__ ebias,
                             int cin, int NKC, int set, int nset) {
    int t = blockIdx.x * blockDim.x + threadIdx.x;
    int total = 2048 * NKC;        // NKC chunks * 4 Mtiles * 32 lanes * 16 halves
    if (t < total) {
        int h     = t & 15;
        int lane  = (t >> 4) & 31;
        int blkid = t >> 9;        // kc*4 + mt
        int mt    = blkid & 3;
        int kc    = blkid >> 2;
        int M     = mt * 16 + (lane & 15);
        int K     = kc * 32 + ((lane >> 4) * 16) + h;
        float s   = gamma[M] * rsqrtf(var[M] + EPSBN);
        float wv  = (K < cin) ? w[(size_t)M * cin + K] * s : 0.f;
        int oblk  = (kc * nset + set) * 4 + mt;
        Wfrag[((size_t)oblk * 32 + lane) * 16 + h] = (_Float16)wv;
    }
    if (t < 64) {
        float s  = gamma[t] * rsqrtf(var[t] + EPSBN);
        ebias[set * 64 + t] = (b[t] - mean[t]) * s + beta[t];
    }
}

extern __shared__ _Float16 lw_dyn[];

// ---------------------------------------------------------------------------
// Single WMMA GEMM (used for p1): Y = relu(W*X + bias), tiled operands.
// ---------------------------------------------------------------------------
template <int KROWS>
__global__ __launch_bounds__(128)
void wmma_gemm64(const _Float16* __restrict__ Wfrag, const float* __restrict__ bias,
                 const _Float16* __restrict__ X, _Float16* __restrict__ Y,
                 int NKC) {
    const int t      = threadIdx.x;
    const int lane   = t & 31;
    const int wave   = t >> 5;
    const int pixblk = (int)blockIdx.x * 8 + wave * 2;
    const int pl     = lane & 15;
    const int khalf  = lane >> 4;

    {
        const uint4* wsrc = (const uint4*)Wfrag;
        uint4*       ldst = (uint4*)lw_dyn;
        const int nvec = NKC * 256;
        for (int i = t; i < nvec; i += 128) ldst[i] = wsrc[i];
    }
    __syncthreads();

    float8 acc0[4] = {};
    float8 acc1[4] = {};

    const unsigned short* xp = (const unsigned short*)X
                               + (size_t)pixblk * (KROWS * 16) + khalf * 256 + pl;
    constexpr int TILE1 = KROWS * 16;
    const half16* lwa = (const half16*)lw_dyn;

    for (int kc = 0; kc < NKC; ++kc) {
        union { half16 h; unsigned int u[8]; } b0, b1;
#pragma unroll
        for (int r = 0; r < 8; ++r) {
            unsigned int lo0 = xp[(2 * r) * 16];
            unsigned int hi0 = xp[(2 * r) * 16 + 16];
            unsigned int lo1 = xp[TILE1 + (2 * r) * 16];
            unsigned int hi1 = xp[TILE1 + (2 * r) * 16 + 16];
            b0.u[r] = lo0 | (hi0 << 16);
            b1.u[r] = lo1 | (hi1 << 16);
        }
        if (kc + 1 < NKC) {
            __builtin_prefetch(xp + 512, 0, 3);
            __builtin_prefetch(xp + TILE1 + 512, 0, 3);
        }
#pragma unroll
        for (int mt = 0; mt < 4; ++mt) {
            half16 afrag = lwa[(kc * 4 + mt) * 32 + lane];
            acc0[mt] = __builtin_amdgcn_wmma_f32_16x16x32_f16(
                false, afrag, false, b0.h, (short)0, acc0[mt], false, false);
            acc1[mt] = __builtin_amdgcn_wmma_f32_16x16x32_f16(
                false, afrag, false, b1.h, (short)0, acc1[mt], false, false);
        }
        xp += 512;
    }

    _Float16* yb = Y + (size_t)pixblk * (64 * 16) + khalf * 128 + pl;
#pragma unroll
    for (int mt = 0; mt < 4; ++mt) {
#pragma unroll
        for (int r = 0; r < 8; ++r) {
            int M    = mt * 16 + r + khalf * 8;
            float bv = bias[M];
            float v0 = fmaxf(acc0[mt][r] + bv, 0.f);
            float v1 = fmaxf(acc1[mt][r] + bv, 0.f);
            yb[(mt * 16 + r) * 16]        = (_Float16)v0;
            yb[1024 + (mt * 16 + r) * 16] = (_Float16)v1;
        }
    }
}

// ---------------------------------------------------------------------------
// Dual WMMA GEMM: two 64-out weight sets (interleaved chunk-major in LDS)
// share ONE pass over X -> 16 WMMAs per K-chunk, halving X HBM traffic,
// which is the dominant stream (3x102MB read once per k+v pair instead of
// twice).  biases: ebias[0..63]=set0, ebias[64..127]=set1.
// ---------------------------------------------------------------------------
template <int KROWS>
__global__ __launch_bounds__(128)
void wmma_gemm_dual(const _Float16* __restrict__ Wfrag, const float* __restrict__ ebias,
                    const _Float16* __restrict__ X,
                    _Float16* __restrict__ Y0, _Float16* __restrict__ Y1,
                    int NKC) {
    const int t      = threadIdx.x;
    const int lane   = t & 31;
    const int wave   = t >> 5;
    const int pixblk = (int)blockIdx.x * 8 + wave * 2;
    const int pl     = lane & 15;
    const int khalf  = lane >> 4;

    {   // stage both interleaved weight images (<=160KB)
        const uint4* wsrc = (const uint4*)Wfrag;
        uint4*       ldst = (uint4*)lw_dyn;
        const int nvec = NKC * 512;
        for (int i = t; i < nvec; i += 128) ldst[i] = wsrc[i];
    }
    __syncthreads();

    float8 aK0[4] = {}, aK1[4] = {};
    float8 aV0[4] = {}, aV1[4] = {};

    const unsigned short* xp = (const unsigned short*)X
                               + (size_t)pixblk * (KROWS * 16) + khalf * 256 + pl;
    constexpr int TILE1 = KROWS * 16;
    const half16* lwa = (const half16*)lw_dyn;

    for (int kc = 0; kc < NKC; ++kc) {
        union { half16 h; unsigned int u[8]; } b0, b1;
#pragma unroll
        for (int r = 0; r < 8; ++r) {
            unsigned int lo0 = xp[(2 * r) * 16];
            unsigned int hi0 = xp[(2 * r) * 16 + 16];
            unsigned int lo1 = xp[TILE1 + (2 * r) * 16];
            unsigned int hi1 = xp[TILE1 + (2 * r) * 16 + 16];
            b0.u[r] = lo0 | (hi0 << 16);
            b1.u[r] = lo1 | (hi1 << 16);
        }
        if (kc + 1 < NKC) {
            __builtin_prefetch(xp + 512, 0, 3);
            __builtin_prefetch(xp + TILE1 + 512, 0, 3);
        }
#pragma unroll
        for (int mt = 0; mt < 4; ++mt) {
            half16 ak = lwa[((kc * 2 + 0) * 4 + mt) * 32 + lane];
            aK0[mt] = __builtin_amdgcn_wmma_f32_16x16x32_f16(
                false, ak, false, b0.h, (short)0, aK0[mt], false, false);
            aK1[mt] = __builtin_amdgcn_wmma_f32_16x16x32_f16(
                false, ak, false, b1.h, (short)0, aK1[mt], false, false);
            half16 av = lwa[((kc * 2 + 1) * 4 + mt) * 32 + lane];
            aV0[mt] = __builtin_amdgcn_wmma_f32_16x16x32_f16(
                false, av, false, b0.h, (short)0, aV0[mt], false, false);
            aV1[mt] = __builtin_amdgcn_wmma_f32_16x16x32_f16(
                false, av, false, b1.h, (short)0, aV1[mt], false, false);
        }
        xp += 512;
    }

    _Float16* y0 = Y0 + (size_t)pixblk * 1024 + khalf * 128 + pl;
    _Float16* y1 = Y1 + (size_t)pixblk * 1024 + khalf * 128 + pl;
#pragma unroll
    for (int mt = 0; mt < 4; ++mt) {
#pragma unroll
        for (int r = 0; r < 8; ++r) {
            int M    = mt * 16 + r + khalf * 8;
            float bk = ebias[M];
            float bv = ebias[64 + M];
            float k0 = fmaxf(aK0[mt][r] + bk, 0.f);
            float k1 = fmaxf(aK1[mt][r] + bk, 0.f);
            float v0 = fmaxf(aV0[mt][r] + bv, 0.f);
            float v1 = fmaxf(aV1[mt][r] + bv, 0.f);
            y0[(mt * 16 + r) * 16]        = (_Float16)k0;
            y0[1024 + (mt * 16 + r) * 16] = (_Float16)k1;
            y1[(mt * 16 + r) * 16]        = (_Float16)v0;
            y1[1024 + (mt * 16 + r) * 16] = (_Float16)v1;
        }
    }
}

// ---------------------------------------------------------------------------
// Per-pixel 3-way attention: dots, softmax, value mix -> ff rows 64..127
// Y operands tiled Krows=64; ff tiled Krows=128.
// ---------------------------------------------------------------------------
__global__ void att_fuse(const _Float16* __restrict__ q,  const _Float16* __restrict__ ck,
                         const _Float16* __restrict__ cv, const _Float16* __restrict__ cok,
                         const _Float16* __restrict__ cov, const _Float16* __restrict__ fk,
                         const _Float16* __restrict__ fv, _Float16* __restrict__ Xff) {
    int p = blockIdx.x * blockDim.x + threadIdx.x;
    if (p >= NPIX) return;
    size_t tb = (size_t)(p >> 4) * (64 * 16) + (p & 15);
    float d0 = 0.f, d1 = 0.f, d2 = 0.f;
    for (int c = 0; c < 64; ++c) {
        size_t o = tb + (size_t)c * 16;
        float qv = (float)q[o];
        d0 += qv * (float)ck[o];
        d1 += qv * (float)cok[o];
        d2 += qv * (float)fk[o];
    }
    float m  = fmaxf(d0, fmaxf(d1, d2));
    float e0 = __expf(d0 - m), e1 = __expf(d1 - m), e2 = __expf(d2 - m);
    float inv = 1.0f / (e0 + e1 + e2);
    float w0 = e0 * inv, w1 = e1 * inv, w2 = e2 * inv;
    _Float16* fo = Xff + (size_t)(p >> 4) * (128 * 16) + (size_t)64 * 16 + (p & 15);
    for (int c = 0; c < 64; ++c) {
        size_t o = tb + (size_t)c * 16;
        float av = w0 * (float)cv[o] + w1 * (float)cov[o] + w2 * (float)fv[o];
        fo[(size_t)c * 16] = (_Float16)av;
    }
}

// final tiny conv: out[b][o][hw] = sum_c h1[c][pix]*pw2[o][c] + pb2[o]
template <int OC>
__global__ void head_out(const _Float16* __restrict__ h1, const float* __restrict__ pw2,
                         const float* __restrict__ pb2, float* __restrict__ out) {
    int pix = blockIdx.x * blockDim.x + threadIdx.x;
    if (pix >= NPIX) return;
    size_t tb = (size_t)(pix >> 4) * (64 * 16) + (pix & 15);
    float acc[OC] = {};
    for (int c = 0; c < 64; ++c) {
        float hv = (float)h1[tb + (size_t)c * 16];
#pragma unroll
        for (int o = 0; o < OC; ++o) acc[o] += hv * pw2[o * 64 + c];
    }
    int b = pix / HWP, hw = pix % HWP;
#pragma unroll
    for (int o = 0; o < OC; ++o)
        out[((size_t)(b * OC) + o) * HWP + hw] = acc[o] + pb2[o];
}

// ---------------------------------------------------------------------------
// Host: params flattened in jax tree order (sorted keys):
//   heads: bbox (base 7), hm (base 7+50)
//   blocks per head: ck,cok,cov,cv,fk,fv,p1 (6 fields each: b,beta,gamma,mean,var,w)
//   then pb2, pw2, then q (6 fields).  Head stride = 50.
// Output: hm [2,3,200,200] then bbox [2,8,200,200] (tuple order).
// ---------------------------------------------------------------------------
extern "C" void kernel_launch(void* const* d_in, const int* in_sizes, int n_in,
                              void* d_out, int out_size, void* d_ws, size_t ws_size,
                              hipStream_t stream) {
    (void)in_sizes; (void)n_in; (void)out_size; (void)ws_size;

    const float* center_feats = (const float*)d_in[0];
    const float* corner_feats = (const float*)d_in[1];
    const float* fg_feats     = (const float*)d_in[2];
    const float* center_grids = (const float*)d_in[3];
    const float* corner_grids = (const float*)d_in[4];
    const float* fg_grids     = (const float*)d_in[5];
    const float* base_feats   = (const float*)d_in[6];
    auto P = [&](int i) { return (const float*)d_in[i]; };

    // workspace carve
    char* ws = (char*)d_ws;
    size_t off = 0;
    auto alloc = [&](size_t bytes) -> char* {
        size_t a = (off + 255) & ~(size_t)255;
        off = a + bytes;
        return ws + a;
    };
    _Float16* Xc  = (_Float16*)alloc((size_t)640 * NPIX * 2);
    _Float16* Xco = (_Float16*)alloc((size_t)640 * NPIX * 2);
    _Float16* Xf  = (_Float16*)alloc((size_t)640 * NPIX * 2);
    _Float16* Xff = (_Float16*)alloc((size_t)128 * NPIX * 2);
    // Y slots: per head {q,ck,cv,cok,cov,fk,fv} = 7, two heads, + shared h1
    _Float16* Yb[15];
    for (int i = 0; i < 15; ++i) Yb[i] = (_Float16*)alloc((size_t)64 * NPIX * 2);
    _Float16* Wfd[7];                      // dual weight images (interleaved)
    for (int i = 0; i < 7; ++i) Wfd[i] = (_Float16*)alloc((size_t)2 * 64 * 640 * 2);
    float* Ebd[7];
    for (int i = 0; i < 7; ++i) Ebd[i] = (float*)alloc(128 * sizeof(float));
    _Float16* Wp1 = (_Float16*)alloc((size_t)64 * 128 * 2);
    float*    Ep1 = (float*)alloc(64 * sizeof(float));

    // stage 1: sample+pack (once, shared by both heads)
    sample_pack<<<(NPIX * 9 + 255) / 256, 256, 0, stream>>>(center_feats, center_grids, Xc, 67, 9);
    sample_pack<<<(NPIX * 8 + 255) / 256, 256, 0, stream>>>(corner_feats, corner_grids, Xco, 76, 8);
    sample_pack<<<(NPIX * 9 + 255) / 256, 256, 0, stream>>>(fg_feats, fg_grids, Xf, 67, 9);
    pack_base<<<(64 * NPIX + 255) / 256, 256, 0, stream>>>(base_feats, Xff);

    const int gemm_grid = NPIX / 128;      // 625, exact
    const int BASE_HM = 7 + 50, BASE_BB = 7;

    auto fold = [&](int po, _Float16* wf, float* eb, int cin, int nkc, int set, int nset) {
        fold_bn_frag<<<(2048 * nkc + 255) / 256, 256, 0, stream>>>(
            P(po + 5), P(po + 0), P(po + 2), P(po + 1), P(po + 3), P(po + 4),
            wf, eb, cin, nkc, set, nset);
    };

    // stage 2a: q for both heads in one dual pass over ff (K=64, NKC=2)
    fold(BASE_HM + 44, Wfd[0], Ebd[0], 64, 2, 0, 2);
    fold(BASE_BB + 44, Wfd[0], Ebd[0], 64, 2, 1, 2);
    wmma_gemm_dual<128><<<gemm_grid, 128, 2 * 2 * 4096, stream>>>(
        Wfd[0], Ebd[0], Xff, Yb[0], Yb[7], 2);      // Yq_hm, Yq_bb

    // stage 2b: k+v dual passes (one X read feeds both GEMMs), per head
    struct KV { int koff, voff, cin; const _Float16* X; };
    KV kv[3] = { {0, 18, 621, Xc}, {6, 12, 624, Xco}, {24, 30, 621, Xf} };
    for (int hd = 0; hd < 2; ++hd) {
        const int base = (hd == 0) ? BASE_HM : BASE_BB;
        for (int s = 0; s < 3; ++s) {
            int slot = 1 + hd * 3 + s;                 // Wfd slot 1..6
            fold(base + kv[s].koff, Wfd[slot], Ebd[slot], kv[s].cin, 20, 0, 2);
            fold(base + kv[s].voff, Wfd[slot], Ebd[slot], kv[s].cin, 20, 1, 2);
            _Float16* Yk = Yb[hd * 7 + 1 + 2 * s];     // ck/cok/fk
            _Float16* Yv = Yb[hd * 7 + 2 + 2 * s];     // cv/cov/fv
            wmma_gemm_dual<640><<<gemm_grid, 128, 2 * 20 * 4096, stream>>>(
                Wfd[slot], Ebd[slot], kv[s].X, Yk, Yv, 20);
        }
    }

    // stage 3: per-head attention -> p1 -> output conv
    for (int hd = 0; hd < 2; ++hd) {
        const int base = (hd == 0) ? BASE_HM : BASE_BB;
        _Float16** Yh = &Yb[hd * 7];   // q,ck,cv,cok,cov,fk,fv
        att_fuse<<<(NPIX + 255) / 256, 256, 0, stream>>>(
            Yh[0], Yh[1], Yh[2], Yh[3], Yh[4], Yh[5], Yh[6], Xff);
        fold(base + 36, Wp1, Ep1, 128, 4, 0, 1);
        wmma_gemm64<128><<<gemm_grid, 128, 4 * 4096, stream>>>(Wp1, Ep1, Xff,
                                                               Yb[14], 4);
        const float* pw2 = P(base + 43);
        const float* pb2 = P(base + 42);
        if (hd == 0)
            head_out<3><<<(NPIX + 255) / 256, 256, 0, stream>>>(Yb[14], pw2, pb2,
                                                                (float*)d_out);
        else
            head_out<8><<<(NPIX + 255) / 256, 256, 0, stream>>>(Yb[14], pw2, pb2,
                                                                (float*)d_out + 2 * 3 * HWP);
    }
}
